// GNNClassifier_38886633898683
// MI455X (gfx1250) — compile-verified, matrix operand order checked by reading, not run on hardware
//
#include <hip/hip_runtime.h>

// ---------------------------------------------------------------------------
// Problem constants (from reference)
// ---------------------------------------------------------------------------
#define BSZ   32
#define TT    512
#define INF_  128
#define HH    256
#define NHEAD 4
#define NCLS  8
#define BT    16384       // BSZ*TT
#define ALPHA 0.2f
#define LNEPS 1e-5f

typedef __attribute__((ext_vector_type(16))) _Float16 v16h;
typedef __attribute__((ext_vector_type(8)))  _Float16 v8h;
typedef __attribute__((ext_vector_type(8)))  float    v8f;

// ---------------------------------------------------------------------------
// WMMA helpers (CDNA5 gfx1250: wave32, V_WMMA_F32_16X16X32_F16)
// A-frag (16x32 f16): lane l<16 : row m=l,    K = {k0..k0+7, k0+16..k0+23}
//                     lane l>=16: row m=l-16, K = {k0+8..k0+15, k0+24..k0+31}
// B-frag mirrors A (outer index = column n).  C/D: lane<16 col n=l, vgpr j =
// row j; lane>=16 row j+8.
// ---------------------------------------------------------------------------
__device__ __forceinline__ int kpat(int i, int hs) {
  return (i < 8) ? (i + 8 * hs) : (i + 8 + 8 * hs);
}

__device__ __forceinline__ v8f wmma_f16(v16h a, v16h b, v8f c) {
  return __builtin_amdgcn_wmma_f32_16x16x32_f16(
      /*neg_a=*/false, a, /*neg_b=*/false, b,
      /*c_mod=*/(short)0, c, /*reuse_a=*/false, /*reuse_b=*/false);
}

// Fast fragment load: p points at element (outer, k0) of a K-contiguous,
// 16B-aligned operand.  Exactly two 16-byte loads per lane.
__device__ __forceinline__ v16h frag_ld2(const _Float16* __restrict__ p) {
  int hs = (threadIdx.x >> 4) & 1;
  v8h lo = *(const v8h*)(p + 8 * hs);
  v8h hi = *(const v8h*)(p + 16 + 8 * hs);
  return __builtin_shufflevector(lo, hi, 0, 1, 2, 3, 4, 5, 6, 7,
                                 8, 9, 10, 11, 12, 13, 14, 15);
}

// wave index as a provably-uniform scalar (keeps guards on the SALU path)
__device__ __forceinline__ int wave_id_scalar() {
  return __builtin_amdgcn_readfirstlane((int)(threadIdx.x >> 5));
}

// ---------------------------------------------------------------------------
// WMMA GEMM:  C[M,N] = act( A[M,K] * B^T + bias )
//   A : f16 row-major [M,K], lda = K (mult of 8); M mult of 16, K mult of 32
//   B : f16 row-major [Npad,K] (N-major "weights"), Npad mult of 64 so every
//       wave does exactly 1 M-tile x 4 N-tiles UNCONDITIONALLY (no PHIs, no
//       accumulator copies).  8 waves/block.
// ---------------------------------------------------------------------------
__global__ void wmma_gemm(const _Float16* __restrict__ A, int lda,
                          const _Float16* __restrict__ B, int ldb,
                          const float* __restrict__ bias,
                          float* __restrict__ C, _Float16* __restrict__ C16,
                          int ldc, int M, int N, int K, int act) {
  int gwave = (int)blockIdx.x * 8 + wave_id_scalar();   // SGPR
  int tn = (N + 15) >> 4;          // 16-col tiles (logical)
  int tng = (tn + 3) >> 2;         // groups of 4 tiles (B padded to 64 cols)
  int tm = M >> 4;
  if (gwave >= tm * tng) return;   // scalar branch, whole wave
  int mi = gwave / tng, ng = gwave % tng;
  int lane = threadIdx.x & 31, idx = lane & 15, hs = lane >> 4;

  const _Float16* Ap = A + (long)(mi * 16 + idx) * lda;
  const _Float16* Bp = B + (long)(ng * 64 + idx) * ldb;
  long bstep = (long)16 * ldb;

  v8f zero = {};
  v8f acc0 = zero, acc1 = zero, acc2 = zero, acc3 = zero;

  for (int k0 = 0; k0 < K; k0 += 32) {
    v16h af = frag_ld2(Ap + k0);
    __builtin_prefetch(Ap + k0 + 64, 0, 3);   // speculative: next A K-slice
    acc0 = wmma_f16(af, frag_ld2(Bp + k0),             acc0);
    acc1 = wmma_f16(af, frag_ld2(Bp + bstep     + k0), acc1);
    acc2 = wmma_f16(af, frag_ld2(Bp + bstep * 2 + k0), acc2);
    acc3 = wmma_f16(af, frag_ld2(Bp + bstep * 3 + k0), acc3);
  }

  v8f acc[4] = {acc0, acc1, acc2, acc3};
#pragma unroll
  for (int s = 0; s < 4; ++s) {
    int col = ng * 64 + s * 16 + idx;
    if (col < N) {                          // per-lane store predicate only
      float bv = bias ? bias[col] : 0.f;
#pragma unroll
      for (int j = 0; j < 8; ++j) {
        int row = mi * 16 + hs * 8 + j;
        float v = acc[s][j] + bv;
        if (act == 1) v = fmaxf(v, 0.f);
        long off = (long)row * ldc + col;
        if (C)   C[off]   = v;
        if (C16) C16[off] = (_Float16)v;
      }
    }
  }
}

// ---------------------------------------------------------------------------
// staging kernels
// ---------------------------------------------------------------------------
__global__ void cvt_f16_kernel(const float* __restrict__ in,
                               _Float16* __restrict__ out, long n) {
  long i = (long)blockIdx.x * blockDim.x + threadIdx.x;
  long stride = (long)gridDim.x * blockDim.x;
  for (; i < n; i += stride) out[i] = (_Float16)in[i];
}

// gat_W [4][256][64] -> Wt [4][64][256] (K-contiguous rows)
__global__ void gatw_t_kernel(const float* __restrict__ W, _Float16* __restrict__ Wt) {
  int i = blockIdx.x * blockDim.x + threadIdx.x;
  if (i >= 4 * 64 * 256) return;
  int k = i & 255, r = i >> 8;
  int n = r & 63, h = r >> 6;
  Wt[i] = (_Float16)W[((long)h * 256 + k) * 64 + n];
}

// fc2_w [8][128] -> zero-padded [64][128] f16 (Npad multiple of 64)
__global__ void fc2_pad_kernel(const float* __restrict__ w, _Float16* __restrict__ wp) {
  int i = blockIdx.x * blockDim.x + threadIdx.x;
  if (i >= 64 * 128) return;
  int n = i >> 7, k = i & 127;
  wp[i] = (n < 8) ? (_Float16)w[n * 128 + k] : (_Float16)0.f;
}

// HG f32 [4][BT][64] -> HGt f16 [4][32][64][512]  (T-contiguous rows)
__global__ void hg_t_kernel(const float* __restrict__ HG, _Float16* __restrict__ HGt) {
  long i = (long)blockIdx.x * blockDim.x + threadIdx.x;
  if (i >= 4L * BT * 64) return;
  int t = (int)(i & 511);
  long r = i >> 9;
  int d = (int)(r & 63);
  long hb = r >> 6;
  int b = (int)(hb & 31), h = (int)(hb >> 5);
  HGt[i] = (_Float16)HG[((long)h * BT + (long)b * TT + t) * 64 + d];
}

// qkv f16 [BT][768] -> Vt f16 [32][4][64][512]
__global__ void v_t_kernel(const _Float16* __restrict__ qkv, _Float16* __restrict__ Vt) {
  long i = (long)blockIdx.x * blockDim.x + threadIdx.x;
  if (i >= 32L * 4 * 64 * 512) return;
  int t = (int)(i & 511);
  long r = i >> 9;
  int d = (int)(r & 63);
  long bh = r >> 6;
  int h = (int)(bh & 3), b = (int)(bh >> 2);
  Vt[i] = qkv[((long)b * TT + t) * 768 + 512 + h * 64 + d];
}

// ---------------------------------------------------------------------------
// Persistent GRU scan: one 768-thread / 24-wave workgroup per layer.
// G[32,768] = h @ W_hh^T via WMMA per step (4 tiles/wave), gates in f32.
// Dynamic LDS: h16 16KB | h_f32 32KB | G_f32 96KB = 144KB (<320KB/WGP).
// ---------------------------------------------------------------------------
__global__ void __launch_bounds__(768)
gru_scan(const float* __restrict__ XP,        // [B,T,768] input projections
         const _Float16* __restrict__ Whh,    // [768,256] f16 (N-major)
         const float* __restrict__ bhh,       // [768]
         _Float16* __restrict__ HS16,         // [B,T,256] f16 out
         float* __restrict__ HS32) {          // optional f32 out
  extern __shared__ char smem[];
  _Float16* h16 = (_Float16*)smem;                  // 8192 halfs
  float*    hf  = (float*)(smem + 16384);           // 8192 floats
  float*    G   = (float*)(smem + 16384 + 32768);   // 24576 floats

  int tid = threadIdx.x;
  for (int e = tid; e < 8192; e += 768) { hf[e] = 0.f; h16[e] = (_Float16)0.f; }
  __syncthreads();

  int wave = wave_id_scalar(), lane = tid & 31, idx = lane & 15, hs = lane >> 4;
  int ni0 = wave, ni1 = wave + 24;                  // 48 N-tiles / 24 waves

  const _Float16* B0 = Whh + (long)(ni0 * 16 + idx) * 256;
  const _Float16* B1 = Whh + (long)(ni1 * 16 + idx) * 256;

  for (int t = 0; t < TT; ++t) {
    v8f zero = {};
    v8f a00 = zero, a01 = zero, a10 = zero, a11 = zero;
#pragma unroll
    for (int kk = 0; kk < 8; ++kk) {
      int k0 = kk * 32;
      v16h fa0 = frag_ld2(&h16[idx * 256 + k0]);
      v16h fa1 = frag_ld2(&h16[(16 + idx) * 256 + k0]);
      v16h fb0 = frag_ld2(B0 + k0);
      v16h fb1 = frag_ld2(B1 + k0);
      a00 = wmma_f16(fa0, fb0, a00);
      a10 = wmma_f16(fa1, fb0, a10);
      a01 = wmma_f16(fa0, fb1, a01);
      a11 = wmma_f16(fa1, fb1, a11);
    }
#pragma unroll
    for (int j = 0; j < 8; ++j) {
      int r0 = hs * 8 + j, r1 = 16 + hs * 8 + j;
      G[r0 * 768 + ni0 * 16 + idx] = a00[j];
      G[r1 * 768 + ni0 * 16 + idx] = a10[j];
      G[r0 * 768 + ni1 * 16 + idx] = a01[j];
      G[r1 * 768 + ni1 * 16 + idx] = a11[j];
    }
    __syncthreads();

    float hnew[11];
#pragma unroll
    for (int c = 0; c < 11; ++c) {
      int e = tid + c * 768;
      if (e < 8192) {
        int b = e >> 8, j = e & 255;
        long xb = ((long)b * TT + t) * 768;
        float xr = XP[xb + j], xz = XP[xb + 256 + j], xn = XP[xb + 512 + j];
        float hr = G[b * 768 + j]       + bhh[j];
        float hz = G[b * 768 + 256 + j] + bhh[256 + j];
        float hn = G[b * 768 + 512 + j] + bhh[512 + j];
        float r = 1.f / (1.f + __expf(-(xr + hr)));
        float z = 1.f / (1.f + __expf(-(xz + hz)));
        float n = tanhf(xn + r * hn);
        hnew[c] = (1.f - z) * n + z * hf[e];
      }
    }
    __syncthreads();
#pragma unroll
    for (int c = 0; c < 11; ++c) {
      int e = tid + c * 768;
      if (e < 8192) {
        float v = hnew[c];
        hf[e] = v; h16[e] = (_Float16)v;
        int b = e >> 8, j = e & 255;
        long o = ((long)b * TT + t) * 256 + j;
        HS16[o] = (_Float16)v;
        if (HS32) HS32[o] = v;
      }
    }
    __syncthreads();
  }
}

// ---------------------------------------------------------------------------
// GAT logits: f_src/f_dst dots with a-vector halves
// ---------------------------------------------------------------------------
__global__ void gat_f_kernel(const float* __restrict__ HG,     // [4][BT][64]
                             const float* __restrict__ gat_a,  // [4][128]
                             float* __restrict__ fsrc, float* __restrict__ fdst) {
  long i = (long)blockIdx.x * blockDim.x + threadIdx.x;
  if (i >= (long)NHEAD * BT) return;
  int h = (int)(i >> 14);
  const float* hv = HG + i * 64;
  const float* a  = gat_a + h * 128;
  float s = 0.f, d = 0.f;
#pragma unroll 8
  for (int k = 0; k < 64; ++k) { s += hv[k] * a[k]; d += hv[k] * a[64 + k]; }
  fsrc[i] = s; fdst[i] = d;
}

__device__ __forceinline__ float lrelu(float x) { return x > 0.f ? x : ALPHA * x; }

// ---------------------------------------------------------------------------
// GAT attention.  Rank-1 logits: row max = leaky(fs_i + max_j fd_j); P built
// directly in A-fragment layout; P@H via WMMA using T-contiguous HGt.
// ---------------------------------------------------------------------------
__global__ void gat_attn(const float* __restrict__ fsrc, const float* __restrict__ fdst,
                         const _Float16* __restrict__ HGt,   // [4][32][64][512]
                         float* __restrict__ gat_out) {      // [BT][256]
  int b = blockIdx.x >> 2, h = blockIdx.x & 3;
  const float* fs = fsrc + ((long)h * BSZ + b) * TT;
  const float* fd = fdst + ((long)h * BSZ + b) * TT;
  const _Float16* Hb = HGt + ((long)(h * BSZ + b) * 64) * 512;

  __shared__ float red[256];
  float lm = -1e30f;
  for (int t = threadIdx.x; t < TT; t += 256) lm = fmaxf(lm, fd[t]);
  red[threadIdx.x] = lm;
  __syncthreads();
  for (int s = 128; s > 0; s >>= 1) {
    if (threadIdx.x < (unsigned)s)
      red[threadIdx.x] = fmaxf(red[threadIdx.x], red[threadIdx.x + s]);
    __syncthreads();
  }
  float Md = red[0];

  int wave = wave_id_scalar(), lane = threadIdx.x & 31, idx = lane & 15, hs = lane >> 4;
  for (int qt = wave; qt < 32; qt += 8) {
    float fsi = fs[qt * 16 + idx];
    float mi  = lrelu(fsi + Md);
    float z = 0.f;
    for (int j = hs * 256; j < hs * 256 + 256; ++j)
      z += __expf(lrelu(fsi + fd[j]) - mi);
    z += __shfl_xor(z, 16);
    float invz = 1.f / z;

    v8f zf = {};
    v8f o0 = zf, o1 = zf, o2 = zf, o3 = zf;
    for (int kc = 0; kc < 16; ++kc) {
      v16h pf;
#pragma unroll
      for (int i = 0; i < 16; ++i) {
        int j = kc * 32 + kpat(i, hs);
        pf[i] = (_Float16)(__expf(lrelu(fsi + fd[j]) - mi) * invz);
      }
      const _Float16* Vc = Hb + kc * 32;
      o0 = wmma_f16(pf, frag_ld2(Vc + (long)(idx)      * 512), o0);
      o1 = wmma_f16(pf, frag_ld2(Vc + (long)(16 + idx) * 512), o1);
      o2 = wmma_f16(pf, frag_ld2(Vc + (long)(32 + idx) * 512), o2);
      o3 = wmma_f16(pf, frag_ld2(Vc + (long)(48 + idx) * 512), o3);
    }
#pragma unroll
    for (int j = 0; j < 8; ++j) {
      long row = (long)b * TT + qt * 16 + hs * 8 + j;
      float* dst = gat_out + row * 256 + h * 64;
      dst[idx]      = o0[j];
      dst[16 + idx] = o1[j];
      dst[32 + idx] = o2[j];
      dst[48 + idx] = o3[j];
    }
  }
}

// ---------------------------------------------------------------------------
// LayerNorm(residual) -> f16 (wave per row, wave32 shuffle reductions)
// ---------------------------------------------------------------------------
__global__ void ln_kernel(const float* __restrict__ g, const float* __restrict__ r,
                          const float* __restrict__ gamma, const float* __restrict__ beta,
                          _Float16* __restrict__ yh) {
  long row = (long)blockIdx.x * 8 + wave_id_scalar();
  if (row >= BT) return;
  int lane = threadIdx.x & 31;
  const float* gr = g + row * 256;
  const float* rr = r + row * 256;
  float vals[8], sum = 0.f;
#pragma unroll
  for (int c = 0; c < 8; ++c) {
    float v = gr[lane + 32 * c] + rr[lane + 32 * c];
    vals[c] = v; sum += v;
  }
#pragma unroll
  for (int off = 1; off < 32; off <<= 1) sum += __shfl_xor(sum, off);
  float mu = sum * (1.f / 256.f), var = 0.f;
#pragma unroll
  for (int c = 0; c < 8; ++c) { float d = vals[c] - mu; var += d * d; }
#pragma unroll
  for (int off = 1; off < 32; off <<= 1) var += __shfl_xor(var, off);
  float inv = rsqrtf(var * (1.f / 256.f) + LNEPS);
#pragma unroll
  for (int c = 0; c < 8; ++c) {
    int col = lane + 32 * c;
    yh[row * 256 + col] = (_Float16)((vals[c] - mu) * inv * gamma[col] + beta[col]);
  }
}

// ---------------------------------------------------------------------------
// MHA: fused attention per (b,head).  Q.K^T via WMMA, 2-pass softmax, P@V via
// WMMA (C->A re-layout through per-wave LDS tile; V pre-transposed).
// ---------------------------------------------------------------------------
__global__ void mha_attn(const _Float16* __restrict__ qkv,  // [BT][768] f16
                         const _Float16* __restrict__ Vt,   // [32][4][64][512]
                         _Float16* __restrict__ ctx16) {    // [BT][256] f16
  int b = blockIdx.x >> 2, h = blockIdx.x & 3;
  int wave = wave_id_scalar(), lane = threadIdx.x & 31, idx = lane & 15, hs = lane >> 4;
  const _Float16* Qb = qkv + (long)b * TT * 768 + h * 64;
  const _Float16* Kb = Qb + 256;
  const _Float16* Vb = Vt + ((long)(b * NHEAD + h) * 64) * 512;
  __shared__ __align__(16) _Float16 ptile[8][16][40];

  for (int qt = wave; qt < 32; qt += 8) {
    int q0 = qt * 16;
    v16h qf0 = frag_ld2(Qb + (long)(q0 + idx) * 768);
    v16h qf1 = frag_ld2(Qb + (long)(q0 + idx) * 768 + 32);

    // pass 1: row maxima of scaled scores
    float mrow[8];
#pragma unroll
    for (int j = 0; j < 8; ++j) mrow[j] = -1e30f;
    for (int kt = 0; kt < 32; ++kt) {
      v8f s = {};
      s = wmma_f16(qf0, frag_ld2(Kb + (long)(kt * 16 + idx) * 768),      s);
      s = wmma_f16(qf1, frag_ld2(Kb + (long)(kt * 16 + idx) * 768 + 32), s);
#pragma unroll
      for (int j = 0; j < 8; ++j) {
        float v = s[j] * 0.125f;
#pragma unroll
        for (int off = 1; off < 16; off <<= 1) v = fmaxf(v, __shfl_xor(v, off));
        mrow[j] = fmaxf(mrow[j], v);
      }
    }

    // pass 2: exp + P@V (unnormalized), Z accumulated per lane-column
    float zrow[8];
#pragma unroll
    for (int j = 0; j < 8; ++j) zrow[j] = 0.f;
    v8f zf = {};
    v8f o0 = zf, o1 = zf, o2 = zf, o3 = zf;
    for (int kc = 0; kc < 16; ++kc) {
#pragma unroll
      for (int sub = 0; sub < 2; ++sub) {
        int kt = kc * 2 + sub;
        v8f s = {};
        s = wmma_f16(qf0, frag_ld2(Kb + (long)(kt * 16 + idx) * 768),      s);
        s = wmma_f16(qf1, frag_ld2(Kb + (long)(kt * 16 + idx) * 768 + 32), s);
#pragma unroll
        for (int j = 0; j < 8; ++j) {
          float p = __expf(s[j] * 0.125f - mrow[j]);
          zrow[j] += p;
          ptile[wave][hs * 8 + j][sub * 16 + idx] = (_Float16)p;
        }
      }
      __syncthreads();
      v16h pf = frag_ld2(&ptile[wave][idx][0]);
      const _Float16* Vc = Vb + kc * 32;
      o0 = wmma_f16(pf, frag_ld2(Vc + (long)(idx)      * 512), o0);
      o1 = wmma_f16(pf, frag_ld2(Vc + (long)(16 + idx) * 512), o1);
      o2 = wmma_f16(pf, frag_ld2(Vc + (long)(32 + idx) * 512), o2);
      o3 = wmma_f16(pf, frag_ld2(Vc + (long)(48 + idx) * 512), o3);
      __syncthreads();
    }
#pragma unroll
    for (int j = 0; j < 8; ++j) {
#pragma unroll
      for (int off = 1; off < 16; off <<= 1) zrow[j] += __shfl_xor(zrow[j], off);
    }
#pragma unroll
    for (int j = 0; j < 8; ++j) {
      float inv = 1.f / zrow[j];
      long row = (long)b * TT + q0 + hs * 8 + j;
      _Float16* dst = ctx16 + row * 256 + h * 64;
      dst[idx]      = (_Float16)(o0[j] * inv);
      dst[16 + idx] = (_Float16)(o1[j] * inv);
      dst[32 + idx] = (_Float16)(o2[j] * inv);
      dst[48 + idx] = (_Float16)(o3[j] * inv);
    }
  }
}

// ---------------------------------------------------------------------------
// mean over T -> pooled f16 [32,256]
// ---------------------------------------------------------------------------
__global__ void pool_kernel(const float* __restrict__ a, _Float16* __restrict__ p) {
  int i = blockIdx.x * blockDim.x + threadIdx.x;
  if (i >= BSZ * HH) return;
  int b = i >> 8, j = i & 255;
  float s = 0.f;
  for (int t = 0; t < TT; ++t) s += a[((long)b * TT + t) * 256 + j];
  p[i] = (_Float16)(s * (1.f / (float)TT));
}

// ---------------------------------------------------------------------------
// host side
// ---------------------------------------------------------------------------
extern "C" void kernel_launch(void* const* d_in, const int* in_sizes, int n_in,
                              void* d_out, int out_size, void* d_ws, size_t ws_size,
                              hipStream_t stream) {
  (void)in_sizes; (void)n_in; (void)out_size; (void)ws_size;
  const float* x    = (const float*)d_in[0];
  const float* Wih0 = (const float*)d_in[1];
  const float* Whh0 = (const float*)d_in[2];
  const float* bih0 = (const float*)d_in[3];
  const float* bhh0 = (const float*)d_in[4];
  const float* Wih1 = (const float*)d_in[5];
  const float* Whh1 = (const float*)d_in[6];
  const float* bih1 = (const float*)d_in[7];
  const float* bhh1 = (const float*)d_in[8];
  const float* gatW = (const float*)d_in[9];
  const float* gata = (const float*)d_in[10];
  const float* lng  = (const float*)d_in[11];
  const float* lnb  = (const float*)d_in[12];
  const float* inw  = (const float*)d_in[13];
  const float* inb  = (const float*)d_in[14];
  const float* outw = (const float*)d_in[15];
  const float* outb = (const float*)d_in[16];
  const float* fc1w = (const float*)d_in[17];
  const float* fc1b = (const float*)d_in[18];
  const float* fc2w = (const float*)d_in[19];
  const float* fc2b = (const float*)d_in[20];
  float* out = (float*)d_out;

  char* ws = (char*)d_ws;
  size_t off = 0;
  auto alloc = [&](size_t bytes) -> void* {
    void* p = ws + off;
    off += (bytes + 255) & ~(size_t)255;
    return p;
  };
  // f16 buffers
  _Float16* xh    = (_Float16*)alloc((size_t)BT * INF_ * 2);
  _Float16* wih0h = (_Float16*)alloc((size_t)768 * 128 * 2);
  _Float16* whh0h = (_Float16*)alloc((size_t)768 * 256 * 2);
  _Float16* wih1h = (_Float16*)alloc((size_t)768 * 256 * 2);
  _Float16* whh1h = (_Float16*)alloc((size_t)768 * 256 * 2);
  _Float16* inwh  = (_Float16*)alloc((size_t)768 * 256 * 2);
  _Float16* outwh = (_Float16*)alloc((size_t)256 * 256 * 2);
  _Float16* fc1wh = (_Float16*)alloc((size_t)128 * 256 * 2);
  _Float16* fc2wp = (_Float16*)alloc((size_t)64 * 128 * 2);   // zero-padded
  _Float16* gatwt = (_Float16*)alloc((size_t)4 * 64 * 256 * 2);
  _Float16* H0h   = (_Float16*)alloc((size_t)BT * 256 * 2);
  _Float16* H1h   = (_Float16*)alloc((size_t)BT * 256 * 2);
  _Float16* HGt   = (_Float16*)alloc((size_t)4 * BT * 64 * 2);
  _Float16* yh    = (_Float16*)alloc((size_t)BT * 256 * 2);
  _Float16* qkvh  = (_Float16*)alloc((size_t)BT * 768 * 2);
  _Float16* Vt    = (_Float16*)alloc((size_t)BT * 256 * 2);
  _Float16* ctxh  = (_Float16*)alloc((size_t)BT * 256 * 2);
  _Float16* poolh = (_Float16*)alloc((size_t)BSZ * 256 * 2);
  _Float16* hidh  = (_Float16*)alloc((size_t)BSZ * 128 * 2);
  // f32 buffers
  float* XP   = (float*)alloc((size_t)BT * 768 * 4);  // reused by both layers
  float* H1   = (float*)alloc((size_t)BT * 256 * 4);
  float* HG   = (float*)alloc((size_t)4 * BT * 64 * 4);
  float* fsrc = (float*)alloc((size_t)4 * BT * 4);
  float* fdst = (float*)alloc((size_t)4 * BT * 4);
  float* res  = (float*)alloc((size_t)BT * 256 * 4);  // gat_out, then attn_out

  auto cvt = [&](const float* in, _Float16* o, long n) {
    int blocks = (int)((n + 255) / 256); if (blocks > 1024) blocks = 1024;
    cvt_f16_kernel<<<blocks, 256, 0, stream>>>(in, o, n);
  };
  auto gemm = [&](const _Float16* A, int lda, const _Float16* B, int ldb,
                  const float* bias, float* C, _Float16* C16, int ldc,
                  int M, int N, int K, int act) {
    int tn = (N + 15) / 16;
    int waves = (M / 16) * ((tn + 3) / 4);
    int blocks = (waves + 7) / 8;     // 8 waves (256 threads) per block
    wmma_gemm<<<blocks, 256, 0, stream>>>(A, lda, B, ldb, bias, C, C16, ldc,
                                          M, N, K, act);
  };

  // --- f16 staging ---
  cvt(x, xh, (long)BT * INF_);
  cvt(Wih0, wih0h, 768L * 128); cvt(Whh0, whh0h, 768L * 256);
  cvt(Wih1, wih1h, 768L * 256); cvt(Whh1, whh1h, 768L * 256);
  cvt(inw, inwh, 768L * 256);   cvt(outw, outwh, 256L * 256);
  cvt(fc1w, fc1wh, 128L * 256);
  fc2_pad_kernel<<<32, 256, 0, stream>>>(fc2w, fc2wp);
  gatw_t_kernel<<<(4 * 64 * 256 + 255) / 256, 256, 0, stream>>>(gatW, gatwt);

  // --- GRU layer 0 ---
  gemm(xh, 128, wih0h, 128, bih0, XP, nullptr, 768, BT, 768, 128, 0);
  hipFuncSetAttribute((const void*)gru_scan,
                      hipFuncAttributeMaxDynamicSharedMemorySize, 160 * 1024);
  gru_scan<<<1, 768, 147456, stream>>>(XP, whh0h, bhh0, H0h, nullptr);

  // --- GRU layer 1 ---
  gemm(H0h, 256, wih1h, 256, bih1, XP, nullptr, 768, BT, 768, 256, 0);
  gru_scan<<<1, 768, 147456, stream>>>(XP, whh1h, bhh1, H1h, H1);

  // --- GAT head projections: HG[i] = H1 @ gat_W[i] ---
  for (int i = 0; i < NHEAD; ++i)
    gemm(H1h, 256, gatwt + (long)i * 64 * 256, 256, nullptr,
         HG + (long)i * BT * 64, nullptr, 64, BT, 64, 256, 0);
  hg_t_kernel<<<(int)((4L * BT * 64 + 255) / 256), 256, 0, stream>>>(HG, HGt);

  gat_f_kernel<<<(NHEAD * BT + 255) / 256, 256, 0, stream>>>(HG, gata, fsrc, fdst);
  gat_attn<<<BSZ * NHEAD, 256, 0, stream>>>(fsrc, fdst, HGt, res);

  // --- LayerNorm(residual) -> yh f16 ---
  ln_kernel<<<BT / 8, 256, 0, stream>>>(H1, res, lng, lnb, yh);

  // --- MHA ---
  gemm(yh, 256, inwh, 256, inb, nullptr, qkvh, 768, BT, 768, 256, 0);
  v_t_kernel<<<(int)((32L * 4 * 64 * 512 + 255) / 256), 256, 0, stream>>>(qkvh, Vt);
  mha_attn<<<BSZ * NHEAD, 256, 0, stream>>>(qkvh, Vt, ctxh);
  gemm(ctxh, 256, outwh, 256, outb, res, nullptr, 256, BT, 256, 256, 0);

  // --- pool + FC head ---
  pool_kernel<<<(BSZ * HH + 255) / 256, 256, 0, stream>>>(res, poolh);
  gemm(poolh, 256, fc1wh, 256, fc1b, nullptr, hidh, 128, BSZ, 128, 256, 1);
  gemm(hidh, 128, fc2wp, 128, fc2b, out, nullptr, NCLS, BSZ, NCLS, 128, 0);
}
